// SpikingHybridCoreFlow_28020366639240
// MI455X (gfx1250) — matrix operand order repaired and checked by reading, not compiled, not run on hardware
//
#include <hip/hip_runtime.h>
#include <math.h>
#include <stdint.h>

#define B_DIM 256
#define N_DIM 2048
#define C_DIM 8
#define T_CYC 32
#define STEPS (T_CYC + C_DIM - 1) /* 39 */

#define TILE_M 64
#define TILE_N 128
#define TILE_K 32
#define A_ELEMS (TILE_M * TILE_K)          /* 2048 bf16 */
#define B_ELEMS (TILE_N * TILE_K)          /* 4096 bf16 */
#define BUF_ELEMS (A_ELEMS + B_ELEMS)      /* 6144 bf16 = 12 KB */

typedef __bf16 bf16;
typedef bf16  bf16x16 __attribute__((ext_vector_type(16)));
typedef bf16  bf16x8  __attribute__((ext_vector_type(8)));
typedef float f32x8   __attribute__((ext_vector_type(8)));

// ---- CDNA5 async global->LDS copy (ASYNCcnt-tracked), per ISA 10.x / 15.18 ----
__device__ __forceinline__ void async_copy_b128(unsigned lds_byte_addr,
                                                unsigned long long gaddr) {
    asm volatile("global_load_async_to_lds_b128 %0, %1, off"
                 :: "v"(lds_byte_addr), "v"(gaddr)
                 : "memory");
}
__device__ __forceinline__ void wait_async0() {
    asm volatile("s_wait_asynccnt 0x0" ::: "memory");
}

// ---------------- utility: zero 32-bit words ----------------
__global__ void zero_words_kernel(unsigned int* __restrict__ p, long n) {
    long i = (long)blockIdx.x * blockDim.x + threadIdx.x;
    if (i < n) p[i] = 0u;
}

// ---- one-time: W fp32 [c][i][o] -> bf16 transposed Wt [c][o][i] (LDS-tiled) ----
__global__ __launch_bounds__(256)
void transpose_w_kernel(const float* __restrict__ W, bf16* __restrict__ Wt) {
    __shared__ bf16 tile[32][33];
    const int c  = blockIdx.z;
    const int i0 = blockIdx.y * 32;
    const int o0 = blockIdx.x * 32;
    const int tx = threadIdx.x;          // 0..31
    const int ty = threadIdx.y;          // 0..7
    const float* Wc = W + (size_t)c * N_DIM * N_DIM;
    bf16* Wtc = Wt + (size_t)c * N_DIM * N_DIM;
    #pragma unroll
    for (int j = 0; j < 32; j += 8)
        tile[ty + j][tx] = (bf16)Wc[(size_t)(i0 + ty + j) * N_DIM + (o0 + tx)];
    __syncthreads();
    #pragma unroll
    for (int j = 0; j < 32; j += 8)
        Wtc[(size_t)(o0 + ty + j) * N_DIM + (i0 + tx)] = tile[tx][ty + j];
}

// ---------------- input spike train: _uniform_spikes(x, t, 32) ----------------
__global__ void spikes_kernel(const float* __restrict__ x, bf16* __restrict__ sig, int t) {
    int i = blockIdx.x * blockDim.x + threadIdx.x; // over B*N
    float rate = x[i];
    int   Ns   = (int)roundf(rate * (float)T_CYC);
    float val  = 0.0f;
    if (Ns == T_CYC) {
        val = 1.0f;
    } else if (Ns > 0 && t < T_CYC) {
        float spacing = (float)T_CYC / (float)Ns;
        float cyc = (float)t;
        if ((floorf(cyc / spacing) < (float)Ns) && (floorf(fmodf(cyc, spacing)) == 0.0f))
            val = 1.0f;
    }
    sig[i] = (bf16)val;
}

// ---------------- one scan step: GEMM (bf16 WMMA) + spiking update ----------------
__global__ __launch_bounds__(256)
void step_kernel(const bf16* __restrict__ sig0,
                 const bf16* __restrict__ bufPrev,
                 bf16*       __restrict__ bufNext,
                 const bf16* __restrict__ Wt,     // [c][o][i] (pre-transposed bf16)
                 float*      __restrict__ memb,
                 float*      __restrict__ counts,
                 const float* __restrict__ thresholds,
                 int t)
{
    const int c     = blockIdx.z;
    const int bRow0 = blockIdx.y * TILE_M;
    const int n0    = blockIdx.x * TILE_N;
    const int tid   = threadIdx.x;
    const size_t coreOff = (size_t)c * B_DIM * N_DIM;

    const bool active = (t >= c) && (t < T_CYC + c);   // uniform per block

    if (!active) {
        // buf2 = buf (carry through ping-pong); last core still feeds counts
        for (int e = tid; e < TILE_M * TILE_N; e += 256) {
            int r = e >> 7, col = e & 127;
            size_t idx = coreOff + (size_t)(bRow0 + r) * N_DIM + (n0 + col);
            bf16 v = bufPrev[idx];
            bufNext[idx] = v;
            if (c == C_DIM - 1)
                counts[(size_t)(bRow0 + r) * N_DIM + (n0 + col)] += (float)v;
        }
        return;
    }

    const bf16* Ag = (c == 0) ? sig0 : (bufPrev + (size_t)(c - 1) * B_DIM * N_DIM);
    const bf16* Bg = Wt + (size_t)c * N_DIM * N_DIM;

    __shared__ __align__(16) bf16 lds[2][BUF_ELEMS];   // 24 KB double-buffered
    const unsigned ldsBase = (unsigned)(size_t)(&lds[0][0]);

    // cooperative staging: 3 x b128 per thread per K-step, all coalesced
    const int arow = tid >> 2;            // 0..63
    const int acol = (tid & 3) * 8;       // 0,8,16,24
    const bf16* aSrc  = Ag + (size_t)(bRow0 + arow) * N_DIM + acol;
    const bf16* bSrc0 = Bg + (size_t)(n0 + arow) * N_DIM + acol;        // n rows 0..63
    const bf16* bSrc1 = Bg + (size_t)(n0 + 64 + arow) * N_DIM + acol;   // n rows 64..127
    const unsigned aDst  = (unsigned)((arow * TILE_K + acol) * sizeof(bf16));
    const unsigned bDst0 = (unsigned)((A_ELEMS + arow * TILE_K + acol) * sizeof(bf16));
    const unsigned bDst1 = (unsigned)((A_ELEMS + (64 + arow) * TILE_K + acol) * sizeof(bf16));

    const int wave   = tid >> 5;
    const int lane   = tid & 31;
    const int laneLo = lane & 15;
    const int laneHi = lane >> 4;          // 0 or 1
    const int mSub   = wave & 3;           // 4 M-subtiles of 16
    const int nBase  = (wave >> 2) * 4;    // each wave owns 4 N-subtiles of 16

    f32x8 acc[4] = {};

    auto stage = [&](int buf, int k0) {
        const unsigned bo = ldsBase + (unsigned)(buf * BUF_ELEMS * sizeof(bf16));
        async_copy_b128(bo + aDst,  (unsigned long long)(aSrc  + k0));
        async_copy_b128(bo + bDst0, (unsigned long long)(bSrc0 + k0));
        async_copy_b128(bo + bDst1, (unsigned long long)(bSrc1 + k0));
    };

    stage(0, 0);
    wait_async0();
    __syncthreads();

    int cur = 0;
    for (int k0 = 0; k0 < N_DIM; k0 += TILE_K) {
        const int nxt = cur ^ 1;
        if (k0 + TILE_K < N_DIM) stage(nxt, k0 + TILE_K);   // prefetch next tile

        const bf16* base = &lds[cur][0];

        // A fragment: lane row = M, K = laneHi*16 .. +15 (contiguous, matches B)
        const bf16* aB = base + (mSub * 16 + laneLo) * TILE_K + laneHi * 16;
        bf16x8 alo = *(const bf16x8*)(aB);
        bf16x8 ahi = *(const bf16x8*)(aB + 8);
        bf16x16 afrag = __builtin_shufflevector(alo, ahi,
            0, 1, 2, 3, 4, 5, 6, 7, 8, 9, 10, 11, 12, 13, 14, 15);

        #pragma unroll
        for (int s = 0; s < 4; ++s) {
            // B fragment: lane col = N, K = laneHi*16 .. +15 (contiguous, Wt is [n][k])
            const bf16* bB = base + A_ELEMS + ((nBase + s) * 16 + laneLo) * TILE_K + laneHi * 16;
            bf16x8 blo = *(const bf16x8*)(bB);
            bf16x8 bhi = *(const bf16x8*)(bB + 8);
            bf16x16 bfrag = __builtin_shufflevector(blo, bhi,
                0, 1, 2, 3, 4, 5, 6, 7, 8, 9, 10, 11, 12, 13, 14, 15);
            acc[s] = __builtin_amdgcn_wmma_f32_16x16x32_bf16(
                         false, afrag, false, bfrag, (short)0, acc[s], false, false);
        }

        wait_async0();
        __syncthreads();
        cur = nxt;
    }

    // ---- spiking epilogue: memb += delta; fire; subtract threshold; counts ----
    const float thr  = thresholds[c];
    const int   mOff = laneHi * 8;   // C/D layout: M = vgpr + 8*(lane>=16), N = lane%16

    #pragma unroll
    for (int s = 0; s < 4; ++s) {
        const int o = n0 + (nBase + s) * 16 + laneLo;
        #pragma unroll
        for (int r = 0; r < 8; ++r) {
            const int bRow = bRow0 + mSub * 16 + mOff + r;
            size_t idx = coreOff + (size_t)bRow * N_DIM + o;
            float m2 = memb[idx] + acc[s][r];
            bool fired = thr < m2;
            bufNext[idx] = fired ? (bf16)1.0f : (bf16)0.0f;
            memb[idx] = fired ? (m2 - thr) : m2;
            if (c == C_DIM - 1)
                counts[(size_t)bRow * N_DIM + o] += fired ? 1.0f : 0.0f;
        }
    }
}

// ---------------- finalize: out = counts / T ----------------
__global__ void finalize_kernel(const float* __restrict__ counts, float* __restrict__ out) {
    int i = blockIdx.x * blockDim.x + threadIdx.x;
    out[i] = counts[i] * (1.0f / (float)T_CYC);
}

extern "C" void kernel_launch(void* const* d_in, const int* in_sizes, int n_in,
                              void* d_out, int out_size, void* d_ws, size_t ws_size,
                              hipStream_t stream) {
    const float* x   = (const float*)d_in[0];                 // (B, N)
    const float* W   = (const float*)d_in[1];                 // (C, N, N)
    const float* thr = (const float*)d_in[2];                 // (C,)

    // workspace layout (all sizes multiples of 256B, laid out contiguously)
    char* ws = (char*)d_ws;
    size_t off = 0;
    bf16*  Wt    = (bf16*)(ws + off);  off += (size_t)C_DIM * N_DIM * N_DIM * 2;  // 67 MB (L2-resident)
    float* memb  = (float*)(ws + off); off += (size_t)C_DIM * B_DIM * N_DIM * 4;  // 16 MB
    bf16*  bufA  = (bf16*)(ws + off);  off += (size_t)C_DIM * B_DIM * N_DIM * 2;  //  8 MB
    bf16*  bufB  = (bf16*)(ws + off);  off += (size_t)C_DIM * B_DIM * N_DIM * 2;  //  8 MB
    float* cnts  = (float*)(ws + off); off += (size_t)B_DIM * N_DIM * 4;          //  2 MB
    bf16*  sig0  = (bf16*)(ws + off);  off += (size_t)B_DIM * N_DIM * 2;          //  1 MB

    // zero memb + bufA + bufB + counts in one contiguous sweep (they're adjacent)
    long zwords = ((size_t)C_DIM * B_DIM * N_DIM * 4 +
                   (size_t)C_DIM * B_DIM * N_DIM * 2 * 2 +
                   (size_t)B_DIM * N_DIM * 4) / 4;
    zero_words_kernel<<<(zwords + 255) / 256, 256, 0, stream>>>((unsigned int*)memb, zwords);

    // one-time: W -> bf16, transposed to [c][o][i]
    dim3 tgrid(N_DIM / 32, N_DIM / 32, C_DIM);
    transpose_w_kernel<<<tgrid, dim3(32, 8), 0, stream>>>(W, Wt);

    dim3 grid(N_DIM / TILE_N, B_DIM / TILE_M, C_DIM);   // (16, 4, 8)
    const int sblocks = (B_DIM * N_DIM) / 256;

    for (int t = 0; t < STEPS; ++t) {
        spikes_kernel<<<sblocks, 256, 0, stream>>>(x, sig0, t);
        const bf16* bp = (t & 1) ? bufB : bufA;
        bf16*       bn = (t & 1) ? bufA : bufB;
        step_kernel<<<grid, 256, 0, stream>>>(sig0, bp, bn, Wt, memb, cnts, thr, t);
    }

    finalize_kernel<<<sblocks, 256, 0, stream>>>(cnts, (float*)d_out);
}
